// SemlaSelfAttention_80161269612631
// MI455X (gfx1250) — matrix operand order repaired and checked by reading, not compiled, not run on hardware
//
#include <hip/hip_runtime.h>

#define NB   1024           // B*N rows
#define K1F  320            // gemm1 reduced K (pf features 128..447)
#define W1STR 328
#define W2STR 264
#define PFSTR 328
#define ACTSTR 264
#define MSGSTR 104

typedef __attribute__((ext_vector_type(16))) __bf16 v16bf;
typedef __attribute__((ext_vector_type(8)))  float  v8f;

union ABfrag { unsigned int u[8]; v16bf v; };

__device__ __forceinline__ unsigned short f2bf(float f) {
  union { float f; unsigned u; } x; x.f = f;
  unsigned r = 0x7FFFu + ((x.u >> 16) & 1u);
  return (unsigned short)((x.u + r) >> 16);
}

// LDS byte offset of a __shared__ object: for the shared aperture the low 32
// bits of the generic address are the LDS offset (CDNA5 ISA 10.2 aperture calc).
__device__ __forceinline__ unsigned ldsOff(const void* p) {
  return (unsigned)(size_t)p;
}

// async copy: global -> LDS, 16 bytes per lane, tracked by ASYNCcnt
__device__ __forceinline__ void asyncCopy16(unsigned ldsDst, const void* gSrc) {
  asm volatile("global_load_async_to_lds_b128 %0, %1, off"
               :: "v"(ldsDst), "v"(gSrc) : "memory");
}
__device__ __forceinline__ void waitAsync0() {
  asm volatile("s_wait_asynccnt 0x0" ::: "memory");
}

// ---------------- small precompute kernels ----------------

// q_msgs (f32) and k_msgs (bf16): invs @ {qW,kW} + {qb,kb}
__global__ __launch_bounds__(128) void k_qkmsg(
    const float* __restrict__ invs, const float* __restrict__ qW,
    const float* __restrict__ qb, const float* __restrict__ kW,
    const float* __restrict__ kb, float* __restrict__ qm,
    unsigned short* __restrict__ km16) {
  __shared__ float inv[256];
  int row = blockIdx.x, t = threadIdx.x;
  inv[t] = invs[row * 256 + t];
  inv[t + 128] = invs[row * 256 + t + 128];
  __syncthreads();
  float sq = qb[t], sk = kb[t];
  for (int f = 0; f < 256; ++f) {
    float iv = inv[f];
    sq += iv * qW[f * 128 + t];
    sk += iv * kW[f * 128 + t];
  }
  qm[row * 128 + t] = sq;
  km16[row * 128 + t] = f2bf(sk);
}

// c_init = b1 + q_msgs @ W1[0:128,:]
__global__ __launch_bounds__(256) void k_cinit(
    const float* __restrict__ qm, const float* __restrict__ W1,
    const float* __restrict__ b1, float* __restrict__ cinit) {
  __shared__ float q[128];
  int row = blockIdx.x, t = threadIdx.x;
  if (t < 128) q[t] = qm[row * 128 + t];
  __syncthreads();
  float s = b1[t];
  for (int f = 0; f < 128; ++f) s += q[f] * W1[f * 256 + t];
  cinit[row * 256 + t] = s;
}

// head_feats = invs @ inW + inb
__global__ __launch_bounds__(256) void k_hf(
    const float* __restrict__ invs, const float* __restrict__ inW,
    const float* __restrict__ inb, float* __restrict__ hf) {
  __shared__ float inv[256];
  int row = blockIdx.x, t = threadIdx.x;
  inv[t] = invs[row * 256 + t];
  __syncthreads();
  float s = inb[t];
  for (int f = 0; f < 256; ++f) s += inv[f] * inW[f * 256 + t];
  hf[row * 256 + t] = s;
}

// proj_equi = equis @ coordW
__global__ __launch_bounds__(192) void k_pe(
    const float* __restrict__ equis, const float* __restrict__ coordW,
    float* __restrict__ pe) {
  __shared__ float eq[192];
  int row = blockIdx.x, t = threadIdx.x;
  eq[t] = equis[row * 192 + t];
  __syncthreads();
  int i = t >> 6, c = t & 63;
  float s = 0.f;
  for (int e = 0; e < 64; ++e) s += eq[i * 64 + e] * coordW[e * 64 + c];
  pe[row * 192 + t] = s;
}

// pack W1[128:448] and W2 to bf16 column-major padded panels
__global__ __launch_bounds__(256) void k_pack(
    const float* __restrict__ W1, const float* __restrict__ W2,
    unsigned short* __restrict__ W1p, unsigned short* __restrict__ W2p) {
  int id = blockIdx.x * 256 + threadIdx.x;
  if (id < 256 * 320) {
    int n = id / 320, k = id - n * 320;
    W1p[n * W1STR + k] = f2bf(W1[(128 + k) * 256 + n]);
  } else {
    int j = id - 256 * 320;
    int n = j / 256, k = j - n * 256;
    W2p[n * W2STR + k] = f2bf(W2[k * 224 + n]);
  }
}

// ---------------- main fused kernel ----------------
// one block per (b,q); 8 waves; online softmax over k; WMMA MLP;
// async global->LDS for weight staging and edge-tile double pump.
__global__ __launch_bounds__(256) void k_main(
    const float* __restrict__ equis, const float* __restrict__ edges,
    const int* __restrict__ adj, const float* __restrict__ b2,
    const unsigned short* __restrict__ wsW1p,
    const unsigned short* __restrict__ wsW2p,
    const unsigned short* __restrict__ wsKm,
    const float* __restrict__ wsCinit, const float* __restrict__ wsHf,
    const float* __restrict__ wsPe, float* __restrict__ wsAo,
    float* __restrict__ wsEo, float* __restrict__ outEdge) {
  __shared__ __align__(16) unsigned short lW1[256 * W1STR];   // 167936 B
  __shared__ __align__(16) unsigned short lW2[224 * W2STR];   // 118272 B
  __shared__ __align__(16) unsigned short lpf[16 * PFSTR];    //  10496 B
  __shared__ __align__(16) unsigned short lact[16 * ACTSTR];  //   8448 B
  __shared__ __align__(16) float ledge[16 * 128];             //   8192 B
  __shared__ float lmsg[16 * MSGSTR];                         //   6656 B
  __shared__ float lcinit[256];
  __shared__ float lqequi[192];
  __shared__ float ladjb[16];

  const int tid = threadIdx.x;
  const int lane = tid & 31;
  const int wv = tid >> 5;
  const int row = blockIdx.x;  // b*256+q
  const int b = row >> 8;
  const int m16 = lane & 15;
  const int kh = lane >> 4;

  // ---- async stage packed weights into LDS (L2 -> LDS, no VGPR roundtrip)
  {
    const unsigned dW1 = ldsOff(lW1);
    const char* g1 = (const char*)wsW1p;
    for (int i = tid * 16; i < 256 * W1STR * 2; i += 256 * 16)
      asyncCopy16(dW1 + i, g1 + i);
    const unsigned dW2 = ldsOff(lW2);
    const char* g2 = (const char*)wsW2p;
    for (int i = tid * 16; i < 224 * W2STR * 2; i += 256 * 16)
      asyncCopy16(dW2 + i, g2 + i);
  }
  // ---- async prefetch edge tile 0 (HBM -> LDS), 8KB contiguous
  const unsigned dE = ldsOff(ledge);
  {
    const char* src = (const char*)(edges + (size_t)(row * 256) * 128);
    asyncCopy16(dE + tid * 16, src + tid * 16);
    asyncCopy16(dE + tid * 16 + 4096, src + tid * 16 + 4096);
  }

  lcinit[tid] = wsCinit[row * 256 + tid];
  if (tid < 192) lqequi[tid] = equis[row * 192 + tid];

  // online-softmax state (threads 0..95: 32 inv heads, 64 equi channels)
  float smax = -1e30f, ssum = 0.f, ssq = 0.f;
  float A0 = 0, A1 = 0, A2 = 0, A3 = 0, A4 = 0, A5 = 0, A6 = 0, A7 = 0;
  __syncthreads();

  for (int kt = 0; kt < 16; ++kt) {
    // ---- retire this iteration's async writes (weights on kt==0, edge tile)
    waitAsync0();
    __syncthreads();

    // ---- build pf tile (16 k-rows x 320 feats, bf16) + adj bias ----
    if (tid < 16) {
      int k = kt * 16 + tid;
      ladjb[tid] = (adj[row * 256 + k] > 0) ? 0.f : -1e30f;
    }
    for (int idx = tid; idx < 16 * K1F; idx += 256) {
      int m = idx / K1F;
      int f = idx - m * K1F;
      int k = kt * 16 + m;
      int rowk = b * 256 + k;
      unsigned short v;
      if (f < 128) {
        v = wsKm[rowk * 128 + f];                       // k_msgs (pre-bf16)
      } else if (f < 192) {                             // dotprods channel
        int c = f - 128;
        const float* ek = equis + rowk * 192;
        float s = lqequi[c] * ek[c] + lqequi[64 + c] * ek[64 + c] +
                  lqequi[128 + c] * ek[128 + c];
        v = f2bf(s);
      } else {                                          // edges (from LDS)
        v = f2bf(ledge[m * 128 + (f - 192)]);
      }
      lpf[m * PFSTR + f] = v;
    }
    __syncthreads();

    // ---- async prefetch next edge tile; retires under the two GEMMs ----
    if (kt + 1 < 16) {
      const char* src =
          (const char*)(edges + (size_t)(row * 256 + (kt + 1) * 16) * 128);
      asyncCopy16(dE + tid * 16, src + tid * 16);
      asyncCopy16(dE + tid * 16 + 4096, src + tid * 16 + 4096);
    }

    // ---- GEMM1: (16 x 320) @ (320 x 256), wave handles 2 col tiles ----
    for (int t2 = 0; t2 < 2; ++t2) {
      int tile = wv * 2 + t2;
      int col = tile * 16 + m16;
      float ci = lcinit[col];  // q-part of pf @ W1 + b1, precomputed
      v8f acc = {ci, ci, ci, ci, ci, ci, ci, ci};
      for (int kk = 0; kk < 10; ++kk) {
        ABfrag A, Bm;
        int ab = m16 * PFSTR + kk * 32 + kh * 8;
#pragma unroll
        for (int v2 = 0; v2 < 8; ++v2) {
          int off = ab + ((v2 >= 4) ? 16 : 0) + (v2 & 3) * 2;
          A.u[v2] = *(const unsigned int*)&lpf[off];
        }
        int bb = col * W1STR + kk * 32 + kh * 8;
#pragma unroll
        for (int v2 = 0; v2 < 8; ++v2)
          Bm.u[v2] = *(const unsigned int*)&lW1[bb + ((v2 >= 4) ? 16 : 0) + (v2 & 3) * 2];
        acc = __builtin_amdgcn_wmma_f32_16x16x32_bf16(false, A.v, false, Bm.v,
                                                      (short)0, acc, false,
                                                      false);
      }
#pragma unroll
      for (int v2 = 0; v2 < 8; ++v2) {  // silu -> act (bf16)
        float x = acc[v2];
        float s = x / (1.f + __expf(-x));
        lact[(v2 + kh * 8) * ACTSTR + col] = f2bf(s);
      }
    }
    __syncthreads();

    // ---- GEMM2: (16 x 256) @ (256 x 224) ----
    for (int t2 = 0; t2 < 2; ++t2) {
      int tile = wv + t2 * 8;
      if (tile >= 14) break;  // wave-uniform
      int col = tile * 16 + m16;
      float bi = b2[col];
      v8f acc = {bi, bi, bi, bi, bi, bi, bi, bi};
      for (int kk = 0; kk < 8; ++kk) {
        ABfrag A, Bm;
        int ab = m16 * ACTSTR + kk * 32 + kh * 8;
#pragma unroll
        for (int v2 = 0; v2 < 8; ++v2) {
          int off = ab + ((v2 >= 4) ? 16 : 0) + (v2 & 3) * 2;
          A.u[v2] = *(const unsigned int*)&lact[off];
        }
        int bb = col * W2STR + kk * 32 + kh * 8;
#pragma unroll
        for (int v2 = 0; v2 < 8; ++v2)
          Bm.u[v2] = *(const unsigned int*)&lW2[bb + ((v2 >= 4) ? 16 : 0) + (v2 & 3) * 2];
        acc = __builtin_amdgcn_wmma_f32_16x16x32_bf16(false, A.v, false, Bm.v,
                                                      (short)0, acc, false,
                                                      false);
      }
      if (col < 96) {  // inv_m + equi_m scores -> LDS
#pragma unroll
        for (int v2 = 0; v2 < 8; ++v2)
          lmsg[(v2 + kh * 8) * MSGSTR + col] = acc[v2];
      } else {  // edge_out straight to HBM
#pragma unroll
        for (int v2 = 0; v2 < 8; ++v2) {
          int k = kt * 16 + v2 + kh * 8;
          outEdge[(row * 256 + k) * 128 + (col - 96)] = acc[v2];
        }
      }
    }
    __syncthreads();

    // ---- online softmax + attention accumulation ----
    if (tid < 96) {
      float x[16], e[16];
      float tmax = -1e30f;
#pragma unroll
      for (int m = 0; m < 16; ++m) {
        x[m] = lmsg[m * MSGSTR + tid] + ladjb[m];
        tmax = fmaxf(tmax, x[m]);
      }
      float nmax = fmaxf(smax, tmax);
      float scale = (smax > -1e29f) ? __expf(smax - nmax) : 0.f;
      float esum = 0.f, esq = 0.f;
#pragma unroll
      for (int m = 0; m < 16; ++m) {
        float ev = (ladjb[m] < -1e29f || nmax < -1e29f)
                       ? 0.f
                       : __expf(x[m] - nmax);
        e[m] = ev;
        esum += ev;
        esq += ev * ev;
      }
      ssum = ssum * scale + esum;
      ssq = ssq * scale * scale + esq;
      smax = nmax;
      if (tid < 32) {  // inv head h: accumulate e * head_feats[k,h,0:8]
        int h = tid;
        float a0 = A0 * scale, a1 = A1 * scale, a2 = A2 * scale,
              a3 = A3 * scale, a4 = A4 * scale, a5 = A5 * scale,
              a6 = A6 * scale, a7 = A7 * scale;
#pragma unroll
        for (int m = 0; m < 16; ++m) {
          const float* hf = wsHf + (b * 256 + kt * 16 + m) * 256 + h * 8;
          float ev = e[m];
          a0 += ev * hf[0]; a1 += ev * hf[1]; a2 += ev * hf[2];
          a3 += ev * hf[3]; a4 += ev * hf[4]; a5 += ev * hf[5];
          a6 += ev * hf[6]; a7 += ev * hf[7];
        }
        A0 = a0; A1 = a1; A2 = a2; A3 = a3;
        A4 = a4; A5 = a5; A6 = a6; A7 = a7;
      } else {  // equi channel c: accumulate e * proj_equi[k,i,c]
        int c = tid - 32;
        float a0 = A0 * scale, a1 = A1 * scale, a2 = A2 * scale;
#pragma unroll
        for (int m = 0; m < 16; ++m) {
          const float* pe = wsPe + (b * 256 + kt * 16 + m) * 192;
          float ev = e[m];
          a0 += ev * pe[c];
          a1 += ev * pe[64 + c];
          a2 += ev * pe[128 + c];
        }
        A0 = a0; A1 = a1; A2 = a2;
      }
    }
    __syncthreads();
  }

  // finalize: multiply by sqrt(sum a^2)/S = sqrt(T)/S^2
  if (tid < 96) {
    float sc = (ssum > 0.f) ? (sqrtf(ssq) / (ssum * ssum)) : 0.f;
    if (tid < 32) {
      float* dst = wsAo + row * 256 + tid * 8;
      dst[0] = A0 * sc; dst[1] = A1 * sc; dst[2] = A2 * sc; dst[3] = A3 * sc;
      dst[4] = A4 * sc; dst[5] = A5 * sc; dst[6] = A6 * sc; dst[7] = A7 * sc;
    } else {
      int c = tid - 32;
      float* dst = wsEo + row * 192;
      dst[c] = A0 * sc;
      dst[64 + c] = A1 * sc;
      dst[128 + c] = A2 * sc;
    }
  }
}

// ---------------- output projections ----------------
__global__ __launch_bounds__(256) void k_outinv(
    const float* __restrict__ ao, const float* __restrict__ outW,
    const float* __restrict__ outb, float* __restrict__ out) {
  __shared__ float a[256];
  int row = blockIdx.x, t = threadIdx.x;
  a[t] = ao[row * 256 + t];
  __syncthreads();
  float s = outb[t];
  for (int f = 0; f < 256; ++f) s += a[f] * outW[f * 256 + t];
  out[row * 256 + t] = s;
}

__global__ __launch_bounds__(192) void k_outequi(
    const float* __restrict__ eo, const float* __restrict__ attnW,
    float* __restrict__ out) {
  __shared__ float a[192];
  int row = blockIdx.x, t = threadIdx.x;
  a[t] = eo[row * 192 + t];
  __syncthreads();
  int i = t >> 6, c = t & 63;
  float s = 0.f;
  for (int e = 0; e < 64; ++e) s += a[i * 64 + e] * attnW[e * 64 + c];
  out[row * 192 + t] = s;
}

// ---------------- launch ----------------
extern "C" void kernel_launch(void* const* d_in, const int* in_sizes, int n_in,
                              void* d_out, int out_size, void* d_ws,
                              size_t ws_size, hipStream_t stream) {
  const float* equis  = (const float*)d_in[0];
  const float* invs   = (const float*)d_in[1];
  const float* edges  = (const float*)d_in[2];
  const int*   adj    = (const int*)d_in[3];
  const float* qW     = (const float*)d_in[4];
  const float* qb     = (const float*)d_in[5];
  const float* kW     = (const float*)d_in[6];
  const float* kb     = (const float*)d_in[7];
  const float* W1     = (const float*)d_in[8];
  const float* b1     = (const float*)d_in[9];
  const float* W2     = (const float*)d_in[10];
  const float* b2     = (const float*)d_in[11];
  const float* inW    = (const float*)d_in[12];
  const float* inb    = (const float*)d_in[13];
  const float* outW   = (const float*)d_in[14];
  const float* outb   = (const float*)d_in[15];
  const float* coordW = (const float*)d_in[16];
  const float* attnW  = (const float*)d_in[17];
  float* out = (float*)d_out;

  char* ws = (char*)d_ws;
  float*          wsQm    = (float*)(ws + 0);                 // 1024*128 f32
  unsigned short* wsKm    = (unsigned short*)(ws + 524288);   // 1024*128 bf16
  float*          wsCinit = (float*)(ws + 786432);            // 1024*256 f32
  float*          wsHf    = (float*)(ws + 1835008);           // 1024*256 f32
  float*          wsPe    = (float*)(ws + 2883584);           // 1024*192 f32
  unsigned short* wsW1p   = (unsigned short*)(ws + 3670016);  // 256*328 bf16
  unsigned short* wsW2p   = (unsigned short*)(ws + 3837952);  // 224*264 bf16
  float*          wsAo    = (float*)(ws + 3956224);           // 1024*256 f32
  float*          wsEo    = (float*)(ws + 5004800);           // 1024*192 f32

  float* outEqui = out;            // (B,N,3,64)
  float* outInv  = out + 196608;   // (B,N,256)
  float* outEdge = out + 458752;   // (B,N,N,128)

  k_qkmsg<<<NB, 128, 0, stream>>>(invs, qW, qb, kW, kb, wsQm, wsKm);
  k_cinit<<<NB, 256, 0, stream>>>(wsQm, W1, b1, wsCinit);
  k_hf<<<NB, 256, 0, stream>>>(invs, inW, inb, wsHf);
  k_pe<<<NB, 192, 0, stream>>>(equis, coordW, wsPe);
  k_pack<<<544, 256, 0, stream>>>(W1, W2, wsW1p, wsW2p);  // 544*256 == 81920+57344
  k_main<<<NB, 256, 0, stream>>>(equis, edges, adj, b2, wsW1p, wsW2p, wsKm,
                                 wsCinit, wsHf, wsPe, wsAo, wsEo, outEdge);
  k_outinv<<<NB, 256, 0, stream>>>(wsAo, outW, outb, outInv);
  k_outequi<<<NB, 192, 0, stream>>>(wsEo, attnW, outEqui);
}
